// MUSE_62500364091757
// MI455X (gfx1250) — compile-verified, matrix operand order checked by reading, not run on hardware
//
#include <hip/hip_runtime.h>
#include <hip/hip_bf16.h>
#include <cstdint>

typedef __attribute__((ext_vector_type(16))) _Float16 v16h;
typedef __attribute__((ext_vector_type(8)))  float    v8f;
typedef __attribute__((ext_vector_type(4)))  unsigned int v4u;
typedef __attribute__((ext_vector_type(8)))  int      v8i;
typedef __attribute__((ext_vector_type(4)))  int      v4i;

#define B_  4096
#define L_  50
#define H_  128
#define BL_ (B_ * L_)   // 204800, divisible by 16

// ---------------- TDM (Tensor Data Mover) support ----------------

#if defined(__has_builtin)
#  if __has_builtin(__builtin_amdgcn_tensor_load_to_lds)
#    define HAVE_TDM 1
#  else
#    define HAVE_TDM 0
#  endif
#else
#  define HAVE_TDM 0
#endif

// ROCm 7.2 (clang-22) builtin takes 5 args; amdgpu-toolchain (clang-23 +
// therock headers, which ship amd_gfx1250_TDM.h) takes 6.
#if __has_include(<hip/amd_detail/amd_gfx1250_TDM.h>)
#  define TDM_6ARG 1
#elif defined(__clang_major__) && (__clang_major__ >= 23)
#  define TDM_6ARG 1
#else
#  define TDM_6ARG 0
#endif

#if defined(__has_builtin)
#  if __has_builtin(__builtin_amdgcn_s_wait_tensorcnt)
#    define TDM_WAIT() __builtin_amdgcn_s_wait_tensorcnt(0)
#  else
#    define TDM_WAIT()
#  endif
#else
#  define TDM_WAIT()
#endif

#if HAVE_TDM
// 1-D contiguous global->LDS DMA. nbytes must be a multiple of 8 (uses 8-byte
// elements, data_size=3). Descriptor per CDNA5 ISA:
// group0 = {count=1, lds_addr, global_addr[56:0], type=2}; group1 packs
// data_size / tensor_dim0 / tensor_dim1=1 / tile_dim0 / stride0; groups 2/3
// zero (dims 2..4 unused for a <=2D tensor).
__device__ __forceinline__ void tdm_load_1d(unsigned int lds_off, const void* gptr,
                                            unsigned int nbytes) {
    unsigned long long ga = (unsigned long long)(uintptr_t)gptr;
    unsigned int ne = nbytes >> 3;                 // 8-byte elements
    v4u g0;
    g0[0] = 1u;                                    // count=1, user descriptor
    g0[1] = lds_off;                               // D#.lds_addr (bytes)
    g0[2] = (unsigned int)(ga & 0xFFFFFFFFu);      // global_addr[31:0]
    g0[3] = (unsigned int)((ga >> 32) & 0x01FFFFFFu) | (2u << 30); // [56:32] | type=2
    v8i g1;
    g1[0] = (3 << 16);                             // wg_mask=0, data_size=3 (8B)
    g1[1] = (int)((ne & 0xFFFFu) << 16);           // tensor_dim0[15:0] at bits 63:48
    g1[2] = (int)((ne >> 16) & 0xFFFFu) | (1 << 16); // tensor_dim0[31:16] | tensor_dim1=1
    g1[3] = (int)((ne & 0xFFFFu) << 16);           // tile_dim0 at bits 127:112
    g1[4] = 0;                                     // tile_dim1/tile_dim2 unused
    g1[5] = (int)ne;                               // tensor_dim0_stride[31:0]
    g1[6] = 0; g1[7] = 0;
    v4i z = {0, 0, 0, 0};
#if TDM_6ARG
    v8i z8 = {0, 0, 0, 0, 0, 0, 0, 0};
    __builtin_amdgcn_tensor_load_to_lds(g0, g1, z, z, z8, 0);
#else
    __builtin_amdgcn_tensor_load_to_lds(g0, g1, z, z, 0);
#endif
}
#endif // HAVE_TDM

// ---------------- WMMA helpers (gfx1250 wave32, 16x16x32 f16 -> f32) ----------------

__device__ __forceinline__ v8f wmma_f16(v16h a, v16h b, v8f c) {
    return __builtin_amdgcn_wmma_f32_16x16x32_f16(false, a, false, b, (short)0, c, false, false);
}

// A/B-fragment from a row-major [rows][K] buffer (row stride = ld); src points
// at [r0][k0]. Row index = lane&15 (M for A, N for B); element e -> K =
// e + (e&8) + 8*(lane>>4) (ISA 7.12.2). Element pairs are K-contiguous, so the
// 16 reads merge into two 16-byte loads per fragment.
__device__ __forceinline__ v16h load_frag_row(const _Float16* src, int ld, int lane) {
    int r = lane & 15, hi = lane >> 4;
    v16h f;
#pragma unroll
    for (int e = 0; e < 16; ++e) {
        int k = e + (e & 8) + 8 * hi;
        f[e] = src[r * ld + k];
    }
    return f;
}

__device__ __forceinline__ float sigf(float x) { return 1.0f / (1.0f + __expf(-x)); }

// ---------------- K0: f32 -> f16 weight conversion ----------------

__global__ void k0_convert(const float* __restrict__ src, _Float16* __restrict__ dst, int n) {
    int i = blockIdx.x * blockDim.x + threadIdx.x;
    if (i < n) dst[i] = (_Float16)src[i];
}

// ---------------- K1: hidden = emb[items]; hv_in = hidden@W_inT+b_in; hv_out likewise ----

__global__ __launch_bounds__(256) void k1_gather_linear(
    const int* __restrict__ items, const float* __restrict__ emb,
    const _Float16* __restrict__ W_in16, const _Float16* __restrict__ W_out16,
    const float* __restrict__ b_in, const float* __restrict__ b_out,
    float* __restrict__ hidden_f32, _Float16* __restrict__ hidden_f16,
    _Float16* __restrict__ hv_in, _Float16* __restrict__ hv_out)
{
    __shared__ __align__(16) _Float16 Atile[16 * H_];
    int row0 = blockIdx.x * 16;
    int tid = threadIdx.x;
    for (int i = tid; i < 16 * H_; i += 256) {
        int r = i >> 7, c = i & 127;
        int row = row0 + r;
        float v = emb[(size_t)items[row] * H_ + c];
        hidden_f32[(size_t)row * H_ + c] = v;
        _Float16 hv = (_Float16)v;
        hidden_f16[(size_t)row * H_ + c] = hv;
        Atile[r * H_ + c] = hv;
    }
    __syncthreads();
    int wave = tid >> 5, lane = tid & 31;
    for (int t = wave; t < 16; t += 8) {          // 8 N-tiles for W_in, 8 for W_out
        const _Float16* W = (t < 8) ? W_in16 : W_out16;
        const float* bias = (t < 8) ? b_in : b_out;
        _Float16* dst = (t < 8) ? hv_in : hv_out;
        int nb = (t & 7) * 16;
        v8f acc = {};
#pragma unroll
        for (int kk = 0; kk < H_; kk += 32) {
            v16h a = load_frag_row(&Atile[kk], H_, lane);
            v16h b = load_frag_row(W + (size_t)nb * H_ + kk, H_, lane); // B[k][n] = W[n][k]
            acc = wmma_f16(a, b, acc);
        }
        int n = nb + (lane & 15);
        float bv = bias[n];
        int hi = lane >> 4;
#pragma unroll
        for (int e = 0; e < 8; ++e)
            dst[(size_t)(row0 + e + 8 * hi) * H_ + n] = (_Float16)(acc[e] + bv);
    }
}

// ---------------- K2: per-session a_in = A_in @ hv_in + b_iah ; a_out likewise ----------
// hv staged TRANSPOSED ([N][K]) so B-fragments are contiguous LDS reads.

__global__ __launch_bounds__(256) void k2_adjacency(
    const float* __restrict__ A, const float* __restrict__ b_iah, const float* __restrict__ b_oah,
    const _Float16* __restrict__ hv_in, const _Float16* __restrict__ hv_out,
    _Float16* __restrict__ inputs)   // [B][L][2H]
{
    __shared__ __align__(16) _Float16 Ain[64 * 64];
    __shared__ __align__(16) _Float16 Aout[64 * 64];
    __shared__ __align__(16) _Float16 HinT[H_ * 64];   // [n][k], zero-padded k
    __shared__ __align__(16) _Float16 HoutT[H_ * 64];
    int b = blockIdx.x;
    int tid = threadIdx.x;
    const float* Ab = A + (size_t)b * L_ * (2 * L_);
    for (int i = tid; i < 64 * 64; i += 256) {    // zero-pad 50 -> 64 in both dims
        int m = i >> 6, k = i & 63;
        _Float16 vi = (_Float16)0.0f, vo = (_Float16)0.0f;
        if (m < L_ && k < L_) {
            vi = (_Float16)Ab[m * (2 * L_) + k];
            vo = (_Float16)Ab[m * (2 * L_) + L_ + k];
        }
        Ain[i] = vi; Aout[i] = vo;
    }
    const _Float16* hin  = hv_in  + (size_t)b * L_ * H_;
    const _Float16* hout = hv_out + (size_t)b * L_ * H_;
    for (int i = tid; i < 64 * H_; i += 256) {    // coalesced global read, strided LDS write
        int k = i >> 7, n = i & 127;
        HinT[n * 64 + k]  = (k < L_) ? hin[k * H_ + n]  : (_Float16)0.0f;
        HoutT[n * 64 + k] = (k < L_) ? hout[k * H_ + n] : (_Float16)0.0f;
    }
    __syncthreads();
    int wave = tid >> 5, lane = tid & 31;
    for (int t = wave; t < 64; t += 8) {          // 2 matrices x 4 Mtiles x 8 Ntiles
        int which = t >> 5;
        int mt = (t >> 3) & 3, nt = t & 7;
        const _Float16* Ap = which ? Aout : Ain;
        const _Float16* Hp = which ? HoutT : HinT;
        const float* bias = which ? b_oah : b_iah;
        v8f acc = {};
#pragma unroll
        for (int kk = 0; kk < 64; kk += 32) {
            v16h a  = load_frag_row(Ap + (mt * 16) * 64 + kk, 64, lane);
            v16h bb = load_frag_row(Hp + (nt * 16) * 64 + kk, 64, lane);
            acc = wmma_f16(a, bb, acc);
        }
        int n = nt * 16 + (lane & 15);
        float bv = bias[n];
        int hi = lane >> 4;
#pragma unroll
        for (int e = 0; e < 8; ++e) {
            int m = mt * 16 + e + 8 * hi;
            if (m < L_)
                inputs[((size_t)b * L_ + m) * (2 * H_) + which * H_ + n] = (_Float16)(acc[e] + bv);
        }
    }
}

// ---------------- K3: GRU cell, fused gi/gh GEMMs + gates, hidden updated in place -------
// The 16x256 input stripe is staged via the Tensor Data Mover (async), overlapped
// with the regular staging of the hidden stripe.

__global__ __launch_bounds__(256) void k3_gru(
    const _Float16* __restrict__ inputs,
    const _Float16* __restrict__ w_ih16, const _Float16* __restrict__ w_hh16,
    const float* __restrict__ b_ih, const float* __restrict__ b_hh,
    float* __restrict__ hidden_f32, _Float16* __restrict__ hidden_f16)
{
    __shared__ __align__(16) _Float16 X[16 * 256];
    __shared__ __align__(16) _Float16 Hh[16 * H_];
    __shared__ float GI[16 * 384];
    __shared__ float GH[16 * 384];
    int row0 = blockIdx.x * 16;
    int tid = threadIdx.x;
#if HAVE_TDM
    if (tid < 32) {   // one wave issues the DMA (TDM ignores EXEC; avoid 8x issue)
        unsigned int lds_off = (unsigned int)(uintptr_t)(void*)&X[0];
        tdm_load_1d(lds_off, inputs + (size_t)row0 * 256, 16 * 256 * sizeof(_Float16));
    }
#else
    for (int i = tid; i < 16 * 256; i += 256) X[i] = inputs[(size_t)row0 * 256 + i];
#endif
    for (int i = tid; i < 16 * H_; i += 256) Hh[i] = hidden_f16[(size_t)row0 * H_ + i];
#if HAVE_TDM
    if (tid < 32) { TDM_WAIT(); }
#endif
    __syncthreads();
    int wave = tid >> 5, lane = tid & 31;
    for (int t = wave; t < 48; t += 8) {          // 24 gi tiles + 24 gh tiles
        bool is_gi = t < 24;
        int nt = is_gi ? t : t - 24;
        const _Float16* W = is_gi ? w_ih16 : w_hh16;
        int K = is_gi ? 256 : H_;
        v8f acc = {};
        for (int kk = 0; kk < K; kk += 32) {
            v16h a = is_gi ? load_frag_row(&X[kk], 256, lane)
                           : load_frag_row(&Hh[kk], H_, lane);
            v16h bb = load_frag_row(W + (size_t)(nt * 16) * K + kk, K, lane);
            acc = wmma_f16(a, bb, acc);
        }
        int n = nt * 16 + (lane & 15);
        float bv = is_gi ? b_ih[n] : b_hh[n];
        float* G = is_gi ? GI : GH;
        int hi = lane >> 4;
#pragma unroll
        for (int e = 0; e < 8; ++e) G[(e + 8 * hi) * 384 + n] = acc[e] + bv;
    }
    __syncthreads();
    for (int i = tid; i < 16 * H_; i += 256) {
        int r = i >> 7, h = i & 127;
        float rg = sigf(GI[r * 384 + h]          + GH[r * 384 + h]);
        float ig = sigf(GI[r * 384 + H_ + h]     + GH[r * 384 + H_ + h]);
        float ng = tanhf(GI[r * 384 + 2 * H_ + h] + rg * GH[r * 384 + 2 * H_ + h]);
        size_t idx = (size_t)(row0 + r) * H_ + h;
        float hv = hidden_f32[idx];
        float out = ng + ig * (hv - ng);
        hidden_f32[idx] = out;
        hidden_f16[idx] = (_Float16)out;
    }
}

// ---------------- K4: seq_hidden = hidden[alias]; ht = seq_hidden[:, len-1] --------------

__global__ void k4_gather(
    const int* __restrict__ alias, const int* __restrict__ lengths,
    const float* __restrict__ hidden_f32,
    float* __restrict__ seq_hidden_out, _Float16* __restrict__ seqh16,
    float* __restrict__ ht, _Float16* __restrict__ ht16)
{
    long gid = (long)blockIdx.x * blockDim.x + threadIdx.x;
    if (gid >= (long)BL_ * H_) return;
    int h = gid & 127;
    int bl = (int)(gid >> 7);
    int b = bl / L_, l = bl - b * L_;
    int src = alias[bl];
    float v = hidden_f32[((size_t)b * L_ + src) * H_ + h];
    seq_hidden_out[gid] = v;
    seqh16[gid] = (_Float16)v;
    int idx = lengths[b] - 1;
    idx = idx < 0 ? 0 : (idx > L_ - 1 ? L_ - 1 : idx);
    if (l == idx) { ht[b * H_ + h] = v; ht16[b * H_ + h] = (_Float16)v; }
}

// ---------------- generic 16-row-stripe GEMM: out = X @ W^T + bias (f32 out) -------------

__global__ __launch_bounds__(256) void k_gemm_bias(
    const _Float16* __restrict__ X, const _Float16* __restrict__ W,
    const float* __restrict__ bias, float* __restrict__ out, int K, int N)
{
    __shared__ __align__(16) _Float16 Xs[16 * 256];
    int row0 = blockIdx.x * 16;
    int tid = threadIdx.x;
#if HAVE_TDM
    if (tid < 32) {
        unsigned int lds_off = (unsigned int)(uintptr_t)(void*)&Xs[0];
        tdm_load_1d(lds_off, X + (size_t)row0 * K, (unsigned int)(16 * K) * sizeof(_Float16));
        TDM_WAIT();
    }
#else
    for (int i = tid; i < 16 * K; i += 256) Xs[i] = X[(size_t)row0 * K + i];
#endif
    __syncthreads();
    int wave = tid >> 5, lane = tid & 31;
    int ntiles = N >> 4;
    for (int t = wave; t < ntiles; t += 8) {
        v8f acc = {};
        for (int kk = 0; kk < K; kk += 32) {
            v16h a = load_frag_row(&Xs[kk], K, lane);
            v16h bb = load_frag_row(W + (size_t)(t * 16) * K + kk, K, lane);
            acc = wmma_f16(a, bb, acc);
        }
        int n = t * 16 + (lane & 15);
        float bv = bias[n];
        int hi = lane >> 4;
#pragma unroll
        for (int e = 0; e < 8; ++e)
            out[(size_t)(row0 + e + 8 * hi) * N + n] = acc[e] + bv;
    }
}

// ---------------- K5: q2 -> sigmoid(q1+q2) -> @W3^T -> masked pooled atomic sum ----------

__global__ __launch_bounds__(256) void k5_attention(
    const _Float16* __restrict__ seqh16, const float* __restrict__ q1,
    const float* __restrict__ b2, const _Float16* __restrict__ W2_16,
    const _Float16* __restrict__ W3_16, const float* __restrict__ b3,
    const unsigned char* __restrict__ mask, float* __restrict__ a_accum)
{
    __shared__ __align__(16) _Float16 Xs[16 * H_];
    __shared__ __align__(16) _Float16 Sg[16 * H_];
    __shared__ float Q2[16 * H_];
    int row0 = blockIdx.x * 16;
    int tid = threadIdx.x;
#if HAVE_TDM
    if (tid < 32) {
        unsigned int lds_off = (unsigned int)(uintptr_t)(void*)&Xs[0];
        tdm_load_1d(lds_off, seqh16 + (size_t)row0 * H_, 16 * H_ * sizeof(_Float16));
        TDM_WAIT();
    }
#else
    for (int i = tid; i < 16 * H_; i += 256) Xs[i] = seqh16[(size_t)row0 * H_ + i];
#endif
    __syncthreads();
    int wave = tid >> 5, lane = tid & 31;
    {   // q2 = seqh @ W2^T + b2 : one 16x16 tile per wave
        int t = wave;
        v8f acc = {};
#pragma unroll
        for (int kk = 0; kk < H_; kk += 32) {
            v16h a = load_frag_row(&Xs[kk], H_, lane);
            v16h bb = load_frag_row(W2_16 + (size_t)(t * 16) * H_ + kk, H_, lane);
            acc = wmma_f16(a, bb, acc);
        }
        int n = t * 16 + (lane & 15);
        float bv = b2[n];
        int hi = lane >> 4;
#pragma unroll
        for (int e = 0; e < 8; ++e) Q2[(e + 8 * hi) * H_ + n] = acc[e] + bv;
    }
    __syncthreads();
    for (int i = tid; i < 16 * H_; i += 256) {    // sigmoid(q1 + q2)
        int r = i >> 7, c = i & 127;
        int b = (row0 + r) / L_;
        Sg[i] = (_Float16)sigf(Q2[i] + q1[b * H_ + c]);
    }
    __syncthreads();
    {   // alp = Sg @ W3^T + b3 ; accumulate alp * seq_hidden * mask into a[b]
        int t = wave;
        v8f acc = {};
#pragma unroll
        for (int kk = 0; kk < H_; kk += 32) {
            v16h a = load_frag_row(&Sg[kk], H_, lane);
            v16h bb = load_frag_row(W3_16 + (size_t)(t * 16) * H_ + kk, H_, lane);
            acc = wmma_f16(a, bb, acc);
        }
        int n = t * 16 + (lane & 15);
        float bv = b3[n];
        int hi = lane >> 4;
#pragma unroll
        for (int e = 0; e < 8; ++e) {
            int r = e + 8 * hi;
            int row = row0 + r;
            int b = row / L_;
            float m = mask[row] ? 1.0f : 0.0f;
            float contrib = (acc[e] + bv) * (float)Xs[r * H_ + n] * m;
            atomicAdd(&a_accum[b * H_ + n], contrib);
        }
    }
}

// ---------------- K6a: concat [a, ht] -> f16 -------------------------------------------

__global__ void k6_concat(const float* __restrict__ a, const _Float16* __restrict__ ht16,
                          _Float16* __restrict__ cat) {
    int i = blockIdx.x * blockDim.x + threadIdx.x;
    if (i >= B_ * 256) return;
    int b = i >> 8, j = i & 255;
    cat[i] = (j < H_) ? (_Float16)a[b * H_ + j] : ht16[b * H_ + (j - H_)];
}

// ---------------- host launch ----------------

extern "C" void kernel_launch(void* const* d_in, const int* in_sizes, int n_in,
                              void* d_out, int out_size, void* d_ws, size_t ws_size,
                              hipStream_t stream) {
    const float* A       = (const float*)d_in[0];
    const int*   items   = (const int*)d_in[1];
    const int*   alias   = (const int*)d_in[2];
    const unsigned char* mask = (const unsigned char*)d_in[3]; // jax bool = 1 byte
    const int*   lengths = (const int*)d_in[4];
    const float* emb     = (const float*)d_in[5];
    const float* W_in = (const float*)d_in[6];  const float* b_in = (const float*)d_in[7];
    const float* W_out= (const float*)d_in[8];  const float* b_out= (const float*)d_in[9];
    const float* w_ih = (const float*)d_in[10]; const float* b_ih = (const float*)d_in[11];
    const float* w_hh = (const float*)d_in[12]; const float* b_hh = (const float*)d_in[13];
    const float* b_iah= (const float*)d_in[14]; const float* b_oah= (const float*)d_in[15];
    const float* W1 = (const float*)d_in[16]; const float* b1 = (const float*)d_in[17];
    const float* W2 = (const float*)d_in[18]; const float* b2 = (const float*)d_in[19];
    const float* W3 = (const float*)d_in[20]; const float* b3 = (const float*)d_in[21];
    const float* Wt = (const float*)d_in[22]; const float* bt = (const float*)d_in[23];

    char* ws = (char*)d_ws;
    size_t off = 0;
    auto take = [&](size_t bytes) -> char* {
        char* p = ws + off;
        off += (bytes + 255) & ~(size_t)255;
        return p;
    };
    _Float16* W_in16  = (_Float16*)take(H_ * H_ * 2);
    _Float16* W_out16 = (_Float16*)take(H_ * H_ * 2);
    _Float16* w_ih16  = (_Float16*)take(384 * 256 * 2);
    _Float16* w_hh16  = (_Float16*)take(384 * H_ * 2);
    _Float16* W1_16   = (_Float16*)take(H_ * H_ * 2);
    _Float16* W2_16   = (_Float16*)take(H_ * H_ * 2);
    _Float16* W3_16   = (_Float16*)take(H_ * H_ * 2);
    _Float16* Wt16    = (_Float16*)take(H_ * 256 * 2);
    float*    hidden_f32 = (float*)   take((size_t)BL_ * H_ * 4);
    _Float16* hidden_f16 = (_Float16*)take((size_t)BL_ * H_ * 2);
    char*     hv_in_raw  = take((size_t)BL_ * H_ * 2);
    char*     hv_out_raw = take((size_t)BL_ * H_ * 2);
    _Float16* inputs16   = (_Float16*)take((size_t)BL_ * 2 * H_ * 2);

    _Float16* hv_in  = (_Float16*)hv_in_raw;
    _Float16* hv_out = (_Float16*)hv_out_raw;
    // after K2 the hv buffers are dead; alias late-stage tensors onto them
    _Float16* seqh16  = (_Float16*)hv_in_raw;                   // 50 MB
    float*    ht      = (float*)   (hv_out_raw);                // 2 MB
    float*    q1      = (float*)   (hv_out_raw + (2u << 20));   // 2 MB
    _Float16* ht16    = (_Float16*)(hv_out_raw + (4u << 20));   // 1 MB
    float*    a_accum = (float*)   (hv_out_raw + (6u << 20));   // 2 MB
    _Float16* cat16   = (_Float16*)(hv_out_raw + (8u << 20));   // 2 MB

    float* out_seq_hidden = (float*)d_out;                      // B*L*H
    float* out_seq_output = (float*)d_out + (size_t)BL_ * H_;   // B*H

    // K0: weight conversions
    auto conv = [&](const float* s, _Float16* d, int n) {
        k0_convert<<<(n + 255) / 256, 256, 0, stream>>>(s, d, n);
    };
    conv(W_in,  W_in16,  H_ * H_);
    conv(W_out, W_out16, H_ * H_);
    conv(w_ih,  w_ih16,  384 * 256);
    conv(w_hh,  w_hh16,  384 * H_);
    conv(W1, W1_16, H_ * H_);
    conv(W2, W2_16, H_ * H_);
    conv(W3, W3_16, H_ * H_);
    conv(Wt, Wt16, H_ * 256);

    // K1: gather + input/output linear projections
    k1_gather_linear<<<BL_ / 16, 256, 0, stream>>>(
        items, emb, W_in16, W_out16, b_in, b_out,
        hidden_f32, hidden_f16, hv_in, hv_out);

    // K2: per-session adjacency matmuls
    k2_adjacency<<<B_, 256, 0, stream>>>(A, b_iah, b_oah, hv_in, hv_out, inputs16);

    // K3: GRU cell (in-place hidden update)
    k3_gru<<<BL_ / 16, 256, 0, stream>>>(
        inputs16, w_ih16, w_hh16, b_ih, b_hh, hidden_f32, hidden_f16);

    // K4: alias gather -> seq_hidden (output) + ht
    {
        long total = (long)BL_ * H_;
        k4_gather<<<(unsigned)((total + 255) / 256), 256, 0, stream>>>(
            alias, lengths, hidden_f32, out_seq_hidden, seqh16, ht, ht16);
    }

    // q1 = ht @ W1^T + b1
    k_gemm_bias<<<B_ / 16, 256, 0, stream>>>(ht16, W1_16, b1, q1, H_, H_);

    // zero the pooled accumulator, then K5 attention
    hipMemsetAsync(a_accum, 0, (size_t)B_ * H_ * 4, stream);
    k5_attention<<<BL_ / 16, 256, 0, stream>>>(
        seqh16, q1, b2, W2_16, W3_16, b3, mask, a_accum);

    // K6: concat + final projection
    k6_concat<<<(B_ * 256) / 256, 256, 0, stream>>>(a_accum, ht16, cat16);
    k_gemm_bias<<<B_ / 16, 256, 0, stream>>>(cat16, Wt16, bt, out_seq_output, 256, H_);
}
